// vector_atnettion_19318762897736
// MI455X (gfx1250) — matrix operand order repaired and checked by reading, not compiled, not run on hardware
//
#include <hip/hip_runtime.h>
#include <math.h>

#define B_    4
#define N_    8192
#define K_    36
#define DIN   64
#define DPRE  67
#define KP    96      // padded reduction dim for pre (3 wmma k-steps of 32)
#define DOUT  128

typedef __attribute__((ext_vector_type(16))) _Float16 v16h;
typedef __attribute__((ext_vector_type(8)))  float    v8f;

union FragAB { v16h v; unsigned int u[8]; };
union FragC  { v8f  v; float f[8]; };

// ---------------------------------------------------------------------------
// Kernel 1: brute-force KNN (top-36 smallest squared distances, per batch).
// One thread per query point; sorted insertion lists in LDS (stride 37 to
// avoid bank conflicts). Candidates packed as float4 -> one broadcast
// ds_load_b128 per candidate in the hot loop.
// ---------------------------------------------------------------------------
#define KNN_T 128
__global__ __launch_bounds__(KNN_T)
void knn_kernel(const float* __restrict__ xyz, int* __restrict__ idx_out) {
    __shared__ float4 cand[KNN_T];
    __shared__ float  td[KNN_T * 37];
    __shared__ int    ti[KNN_T * 37];

    const int t = threadIdx.x;
    const int q = blockIdx.x * KNN_T + t;      // global row in [0, B*N)
    const int b = q >> 13;                     // q / N_
    const int qi = q & (N_ - 1);
    const float* bx = xyz + (size_t)b * N_ * 3;

    const float qx = bx[qi * 3 + 0];
    const float qy = bx[qi * 3 + 1];
    const float qz = bx[qi * 3 + 2];

    float* myd = td + t * 37;
    int*   myi = ti + t * 37;
    for (int j = 0; j < K_; ++j) { myd[j] = 3.4e38f; myi[j] = 0; }
    float worst = 3.4e38f;

    for (int t0 = 0; t0 < N_; t0 += KNN_T) {
        __syncthreads();
        cand[t] = make_float4(bx[(t0 + t) * 3 + 0],
                              bx[(t0 + t) * 3 + 1],
                              bx[(t0 + t) * 3 + 2], 0.f);
        __syncthreads();
        for (int c = 0; c < KNN_T; ++c) {
            const float4 cd = cand[c];
            const float dx = qx - cd.x;
            const float dy = qy - cd.y;
            const float dz = qz - cd.z;
            const float d = dx * dx + dy * dy + dz * dz;
            if (d < worst) {                   // strict < keeps earlier index on ties
                int j = K_ - 1;
                while (j > 0 && myd[j - 1] > d) {
                    myd[j] = myd[j - 1]; myi[j] = myi[j - 1]; --j;
                }
                myd[j] = d; myi[j] = t0 + c;
                worst = myd[K_ - 1];
            }
        }
    }
    int* out = idx_out + (size_t)q * K_;
    for (int j = 0; j < K_; ++j) out[j] = myi[j];
}

// ---------------------------------------------------------------------------
// Kernel 2: P1 = pre@Wr1^T, P2 = pre@Wr2^T + br, VF = relu(pre@Wv^T + bv)
// pre = concat(feature, xyz), padded 67 -> 96. f16 WMMA, f32 accumulate.
// Block = 128 threads (4 waves), 64 rows per block; weights staged one matrix
// at a time in LDS. Epilogue is branchless (select dst/bias, cndmask relu).
// ---------------------------------------------------------------------------
#define PROJ_ROWS 64
__global__ __launch_bounds__(128)
void proj_kernel(const float* __restrict__ feature, const float* __restrict__ xyz,
                 const float* __restrict__ Wr, const float* __restrict__ br,
                 const float* __restrict__ Wv, const float* __restrict__ bv,
                 float* __restrict__ P1, float* __restrict__ P2, float* __restrict__ VF) {
    __shared__ _Float16 preh[PROJ_ROWS][KP];
    __shared__ _Float16 wsh[DOUT][KP];

    const int t    = threadIdx.x;
    const int lane = t & 31;
    const int wave = t >> 5;
    const int r0   = blockIdx.x * PROJ_ROWS;
    const int mrow = lane & 15;
    const int hi4  = lane >> 4;

    // Preload biases for this lane's 8 output columns (hoisted out of hot loop).
    float bias_r[8], bias_v[8];
    #pragma unroll
    for (int ct = 0; ct < 8; ++ct) {
        bias_r[ct] = br[ct * 16 + mrow];
        bias_v[ct] = bv[ct * 16 + mrow];
    }

    // Stage pre tile (f32 -> f16), zero-padded to KP.
    for (int i = t; i < PROJ_ROWS * KP; i += 128) {
        const int rr = i / KP, k = i % KP;
        const int g = r0 + rr;
        float v = 0.f;
        if (k < DIN)       v = feature[(size_t)g * DIN + k];
        else if (k < DPRE) v = xyz[(size_t)g * 3 + (k - DIN)];
        preh[rr][k] = (_Float16)v;
    }
    __syncthreads();

    // A fragments (16-bit A 16x32 layout), reused for all 3 matrices.
    FragAB afrag[3];
    #pragma unroll
    for (int ks = 0; ks < 3; ++ks)
        #pragma unroll
        for (int j = 0; j < 8; ++j) {
            const int k = ks * 32 + (j >> 2) * 16 + hi4 * 8 + (j & 3) * 2;
            afrag[ks].u[j] = *(const unsigned int*)&preh[wave * 16 + mrow][k];
        }

    #pragma unroll
    for (int mat = 0; mat < 3; ++mat) {
        __syncthreads();
        // Stage this matrix's weights as f16 (row-major [out_ch][k], padded).
        for (int i = t; i < DOUT * KP; i += 128) {
            const int c = i / KP, k = i % KP;
            float v = 0.f;
            if (k < DPRE) {
                if (mat == 0)      v = Wr[(size_t)c * (2 * DPRE) + k];
                else if (mat == 1) v = Wr[(size_t)c * (2 * DPRE) + DPRE + k];
                else               v = Wv[(size_t)c * DPRE + k];
            }
            wsh[c][k] = (_Float16)v;
        }
        __syncthreads();

        float* const dst = (mat == 0) ? P1 : (mat == 1) ? P2 : VF;

        for (int ct = 0; ct < 8; ++ct) {
            const int n = ct * 16 + mrow;
            // Load all 3 B fragments up front so ds_loads overlap.
            FragAB bfrag[3];
            #pragma unroll
            for (int ks = 0; ks < 3; ++ks)
                #pragma unroll
                for (int j = 0; j < 8; ++j) {
                    const int k = ks * 32 + (j >> 2) * 16 + hi4 * 8 + (j & 3) * 2;
                    bfrag[ks].u[j] = *(const unsigned int*)&wsh[n][k];
                }
            v8f accv = {};
            accv = __builtin_amdgcn_wmma_f32_16x16x32_f16(
                false, afrag[0].v, false, bfrag[0].v, (short)0, accv, false, false);
            accv = __builtin_amdgcn_wmma_f32_16x16x32_f16(
                false, afrag[1].v, false, bfrag[1].v, (short)0, accv, false, false);
            accv = __builtin_amdgcn_wmma_f32_16x16x32_f16(
                false, afrag[2].v, false, bfrag[2].v, (short)0, accv, false, false);

            const float bias = (mat == 0) ? 0.f : (mat == 1) ? bias_r[ct] : bias_v[ct];
            FragC res; res.v = accv;
            #pragma unroll
            for (int r = 0; r < 8; ++r) {
                const int row = r0 + wave * 16 + hi4 * 8 + r;
                float v = res.f[r] + bias;
                v = (mat == 2) ? fmaxf(v, 0.f) : v;   // cndmask, no branch
                dst[(size_t)row * DOUT + n] = v;
            }
        }
    }
}

// ---------------------------------------------------------------------------
// Kernel 3: fused gather + channel softmax + neighbor reduction + final GEMM.
// Block = 128 threads (4 waves) handles 16 query points.
// Stage 1: thread (p = t/8, s = t%8) owns 16 channels of point p; softmax
//          max/sum reduced across the 8-lane group via __shfl_xor.
// Stage 2: out[16][128] (f16 in LDS) @ Ws^T via WMMA, + bs, stored f32.
// ---------------------------------------------------------------------------
__global__ __launch_bounds__(128)
void attn_kernel(const int* __restrict__ idx, const float* __restrict__ P1,
                 const float* __restrict__ P2, const float* __restrict__ VF,
                 const float* __restrict__ Ws, const float* __restrict__ bs,
                 float* __restrict__ out) {
    __shared__ int      idxs[16][K_];
    __shared__ _Float16 outsh[16][DOUT];
    __shared__ _Float16 wssh[DOUT][DOUT];

    const int t  = threadIdx.x;
    const int r0 = blockIdx.x * 16;            // global row base (one batch only)
    const int b  = r0 >> 13;                   // r0 / N_

    const int lane = t & 31;
    const int wave = t >> 5;
    const int mrow = lane & 15;
    const int hi4  = lane >> 4;

    // Preload final-bias values for this lane's two column tiles.
    float bsv[2];
    #pragma unroll
    for (int i = 0; i < 2; ++i) bsv[i] = bs[(wave * 2 + i) * 16 + mrow];

    for (int i = t; i < DOUT * DOUT; i += 128)
        wssh[i >> 7][i & 127] = (_Float16)Ws[i];
    for (int i = t; i < 16 * K_; i += 128)
        idxs[i / K_][i % K_] = idx[(size_t)(r0 + i / K_) * K_ + i % K_];
    __syncthreads();

    const int p  = t >> 3;
    const int s  = t & 7;
    const int c0 = s * 16;
    const int row_p = r0 + p;

    float p1n[16], p2n[16], acc[16];
    #pragma unroll
    for (int j = 0; j < 16; ++j) {
        p1n[j] = P1[(size_t)row_p * DOUT + c0 + j];
        p2n[j] = P2[(size_t)row_p * DOUT + c0 + j];   // br already folded in
        acc[j] = 0.f;
    }

    for (int k = 0; k < K_; ++k) {
        const int g = b * N_ + idxs[p][k];
        const float* p1g = P1 + (size_t)g * DOUT + c0;
        const float* vfg = VF + (size_t)g * DOUT + c0;
        float lg[16];
        float mx = -3.4e38f;
        #pragma unroll
        for (int j = 0; j < 16; ++j) {
            lg[j] = p1g[j] + p2n[j] - p1n[j];
            mx = fmaxf(mx, lg[j]);
        }
        #pragma unroll
        for (int m = 1; m < 8; m <<= 1) mx = fmaxf(mx, __shfl_xor(mx, m, 32));
        float e[16], sum = 0.f;
        #pragma unroll
        for (int j = 0; j < 16; ++j) { e[j] = __expf(lg[j] - mx); sum += e[j]; }
        #pragma unroll
        for (int m = 1; m < 8; m <<= 1) sum += __shfl_xor(sum, m, 32);
        const float inv = 1.f / sum;
        #pragma unroll
        for (int j = 0; j < 16; ++j) acc[j] += vfg[j] * (e[j] * inv);
    }

    #pragma unroll
    for (int j = 0; j < 16; ++j) outsh[p][c0 + j] = (_Float16)acc[j];
    __syncthreads();

    // Stage 2: final[16][128] = outsh @ Ws^T (+bs)
    #pragma unroll
    for (int i = 0; i < 2; ++i) {
        const int ct = wave * 2 + i;
        v8f accv = {};
        #pragma unroll
        for (int ks = 0; ks < 4; ++ks) {
            FragAB a, bb;
            #pragma unroll
            for (int j = 0; j < 8; ++j) {
                const int ka = ks * 32 + (j >> 2) * 16 + hi4 * 8 + (j & 3) * 2;
                const int kb = ks * 32 + (j >> 2) * 16 + hi4 * 8 + (j & 3) * 2;
                a.u[j]  = *(const unsigned int*)&outsh[mrow][ka];
                bb.u[j] = *(const unsigned int*)&wssh[ct * 16 + mrow][kb];
            }
            accv = __builtin_amdgcn_wmma_f32_16x16x32_f16(
                false, a.v, false, bb.v, (short)0, accv, false, false);
        }
        FragC res; res.v = accv;
        #pragma unroll
        for (int r = 0; r < 8; ++r) {
            const int row = r0 + hi4 * 8 + r;
            const int col = ct * 16 + mrow;
            out[(size_t)row * DOUT + col] = res.f[r] + bsv[i];
        }
    }
}

// ---------------------------------------------------------------------------
extern "C" void kernel_launch(void* const* d_in, const int* in_sizes, int n_in,
                              void* d_out, int out_size, void* d_ws, size_t ws_size,
                              hipStream_t stream) {
    (void)in_sizes; (void)n_in; (void)out_size; (void)ws_size;
    const float* feature = (const float*)d_in[0];
    const float* xyz     = (const float*)d_in[1];
    const float* Wr      = (const float*)d_in[2];
    const float* br      = (const float*)d_in[3];
    const float* Wv      = (const float*)d_in[4];
    const float* bv      = (const float*)d_in[5];
    const float* Ws      = (const float*)d_in[6];
    const float* bs      = (const float*)d_in[7];
    // d_in[8] = knn_num (36, hardcoded as K_)

    char* ws = (char*)d_ws;
    size_t o = 0;
    int*   idx = (int*)(ws + o); o += (size_t)B_ * N_ * K_ * sizeof(int);
    float* P1  = (float*)(ws + o); o += (size_t)B_ * N_ * DOUT * sizeof(float);
    float* P2  = (float*)(ws + o); o += (size_t)B_ * N_ * DOUT * sizeof(float);
    float* VF  = (float*)(ws + o);

    knn_kernel <<<(B_ * N_) / KNN_T, KNN_T, 0, stream>>>(xyz, idx);
    proj_kernel<<<(B_ * N_) / PROJ_ROWS, 128, 0, stream>>>(feature, xyz, Wr, br, Wv, bv,
                                                           P1, P2, VF);
    attn_kernel<<<(B_ * N_) / 16, 128, 0, stream>>>(idx, P1, P2, VF, Ws, bs,
                                                    (float*)d_out);
}
